// Gemma3Attention_6511170421315
// MI455X (gfx1250) — matrix operand order repaired and checked by reading, not compiled, not run on hardware
//
#include <hip/hip_runtime.h>
#include <hip/hip_bf16.h>
#include <math.h>

// ---------- problem constants ----------
constexpr int Sq   = 4096;   // sequence length
constexpr int Dm   = 2048;   // model dim
constexpr int Hq   = 16;     // query heads
constexpr int KVh  = 4;      // kv heads
constexpr int HD   = 128;    // head dim
constexpr int SWIN = 1024;   // sliding window
// SCALE = 256^-0.5 = 0.0625 ; CAP = 50 ; EPS = 1e-6

// ---------- types ----------
typedef __bf16 bf16_t;
typedef __attribute__((ext_vector_type(16))) __bf16 v16bf;
typedef __attribute__((ext_vector_type(8)))  __bf16 v8bf;
typedef __attribute__((ext_vector_type(8)))  float  v8f;

union Frag { v16bf v; v8bf h[2]; };

static __device__ inline v8f wmma_bf16(v16bf a, v16bf b, v8f c) {
    return __builtin_amdgcn_wmma_f32_16x16x32_bf16(false, a, false, b, (short)0, c, false, false);
}

static __device__ inline float half_max16(float v) {
    v = fmaxf(v, __shfl_xor(v, 1, 32));
    v = fmaxf(v, __shfl_xor(v, 2, 32));
    v = fmaxf(v, __shfl_xor(v, 4, 32));
    v = fmaxf(v, __shfl_xor(v, 8, 32));
    return v;
}
static __device__ inline float half_sum16(float v) {
    v += __shfl_xor(v, 1, 32);
    v += __shfl_xor(v, 2, 32);
    v += __shfl_xor(v, 4, 32);
    v += __shfl_xor(v, 8, 32);
    return v;
}

// ---------- conversion kernels ----------
__global__ void cvt_bf16_kernel(const float* __restrict__ in, bf16_t* __restrict__ out, size_t n) {
    size_t stride = (size_t)gridDim.x * blockDim.x;
    for (size_t i = (size_t)blockIdx.x * blockDim.x + threadIdx.x; i < n; i += stride)
        out[i] = (bf16_t)in[i];
}

// W is [K][N] row-major fp32 -> Bt is [N][K] bf16
__global__ void transpose_cvt_kernel(const float* __restrict__ W, bf16_t* __restrict__ Bt,
                                     int K, int N) {
    size_t n = (size_t)K * N;
    size_t stride = (size_t)gridDim.x * blockDim.x;
    for (size_t i = (size_t)blockIdx.x * blockDim.x + threadIdx.x; i < n; i += stride) {
        int k = (int)(i / N);
        int c = (int)(i % N);
        Bt[(size_t)c * K + k] = (bf16_t)W[i];
    }
}

// ---------- head-tile GEMM: block computes C(64 x 128) = A(64 x Ktot) * Bt^T ----------
// 4 waves, each wave owns a 32x64 sub-tile: 2 A-frags x 4 B-frags = 8 WMMA per K-step.
// MODE 0: rmsnorm + rope epilogue -> bf16 out[(h*Srows+row)*128 + c]   (Q / K)
// MODE 1: plain bf16, transposed  -> out[(h*128+c)*Srows + row]        (V)
// MODE 2: plain fp32              -> outf[row*outW + h*128 + c]        (final proj)
template <int MODE>
__global__ __launch_bounds__(128) void gemm_head_kernel(
    const bf16_t* __restrict__ A, const bf16_t* __restrict__ Bt, int Ktot,
    const float* __restrict__ cosp, const float* __restrict__ sinp,
    const float* __restrict__ nw,
    bf16_t* __restrict__ outb, float* __restrict__ outf,
    int Srows, int outW)
{
    const int m0   = blockIdx.x * 64;
    const int h    = blockIdx.y;
    const int tid  = threadIdx.x;
    const int w    = tid >> 5;
    const int mi   = w >> 1;       // which 32-row half of the block tile
    const int ni   = w & 1;        // which 64-col half of the head
    const int lane = tid & 31;
    const int l16  = lane & 15;
    const int half = lane >> 4;

    __shared__ float tile[64][128];
    __shared__ float red[64][2];

    const int ka = half * 8;    // A-frag K offset for this half
    const int kb = half * 16;   // unused: B-frags load with ka, like A

    const bf16_t* arow0 = A + (size_t)(m0 + mi * 32 + l16) * Ktot;        // rows 0..15 of sub-tile
    const bf16_t* arow1 = A + (size_t)(m0 + mi * 32 + 16 + l16) * Ktot;   // rows 16..31
    const bf16_t* brow[4];
#pragma unroll
    for (int t = 0; t < 4; t++)
        brow[t] = Bt + (size_t)(h * 128 + ni * 64 + t * 16 + l16) * Ktot;

    v8f acc[2][4];
#pragma unroll
    for (int i = 0; i < 2; i++)
#pragma unroll
        for (int t = 0; t < 4; t++)
            acc[i][t] = (v8f){0.f,0.f,0.f,0.f,0.f,0.f,0.f,0.f};

    for (int k0 = 0; k0 < Ktot; k0 += 32) {
        Frag a0, a1;
        a0.h[0] = *(const v8bf*)(arow0 + k0 + ka);
        a0.h[1] = *(const v8bf*)(arow0 + k0 + 16 + ka);
        a1.h[0] = *(const v8bf*)(arow1 + k0 + ka);
        a1.h[1] = *(const v8bf*)(arow1 + k0 + 16 + ka);
        Frag b[4];
#pragma unroll
        for (int t = 0; t < 4; t++) {
            b[t].h[0] = *(const v8bf*)(brow[t] + k0 + ka);
            b[t].h[1] = *(const v8bf*)(brow[t] + k0 + 16 + ka);
        }
#pragma unroll
        for (int t = 0; t < 4; t++) {
            acc[0][t] = wmma_bf16(a0.v, b[t].v, acc[0][t]);
            acc[1][t] = wmma_bf16(a1.v, b[t].v, acc[1][t]);
        }
    }

#pragma unroll
    for (int i = 0; i < 2; i++)
#pragma unroll
        for (int t = 0; t < 4; t++)
#pragma unroll
            for (int j = 0; j < 8; j++)
                tile[mi * 32 + i * 16 + j + 8 * half][ni * 64 + t * 16 + l16] = acc[i][t][j];
    __syncthreads();

    // epilogue: 128 threads over 64 rows x 128 cols (2 threads/row, 64 cols each)
    const int r = tid >> 1;
    const int g = tid & 1;
    const int cbase = g * 64;
    const int grow = m0 + r;

    if (MODE == 0) {
        float ss = 0.f;
        for (int i = 0; i < 64; i++) { float x = tile[r][cbase + i]; ss += x * x; }
        red[r][g] = ss;
        __syncthreads();
        ss = red[r][0] + red[r][1];
        float rms = rsqrtf(ss * (1.0f / 128.0f) + 1e-6f);
        for (int i = 0; i < 64; i++) {
            int c = cbase + i;
            int p = (c < 64) ? c + 64 : c - 64;
            float xn  = tile[r][c] * rms * (1.f + nw[c]);
            float xp  = tile[r][p] * rms * (1.f + nw[p]);
            float rot = (c < 64) ? -xp : xp;
            float cs  = cosp[(size_t)grow * 128 + c];
            float sn  = sinp[(size_t)grow * 128 + c];
            outb[((size_t)h * Srows + grow) * 128 + c] = (bf16_t)(xn * cs + rot * sn);
        }
    } else if (MODE == 1) {
        for (int i = 0; i < 64; i++) {
            int c = cbase + i;
            outb[((size_t)(h * 128 + c)) * Srows + grow] = (bf16_t)tile[r][c];
        }
    } else {
        for (int i = 0; i < 64; i++) {
            int c = cbase + i;
            outf[(size_t)grow * outW + h * 128 + c] = tile[r][c];
        }
    }
}

// ---------- flash attention, sliding window + tanh softcap ----------
// Qb: [H][S][HD] bf16 ; Kb: [KV][S][HD] bf16 ; Vt: [KV][HD][S] bf16
// Ab: [S][H*HD]  bf16
__global__ __launch_bounds__(128) void attn_kernel(
    const bf16_t* __restrict__ Qb, const bf16_t* __restrict__ Kb,
    const bf16_t* __restrict__ Vt, bf16_t* __restrict__ Ab)
{
    const int h    = blockIdx.y;
    const int hk   = h >> 2;             // GQA: 4 q heads per kv head
    const int w    = threadIdx.x >> 5;
    const int lane = threadIdx.x & 31;
    const int l16  = lane & 15;
    const int half = lane >> 4;
    const int m0   = (blockIdx.x * 4 + w) * 16;

    __shared__ bf16_t pls[4][16][32];    // per-wave P tile (C-layout -> A-layout bounce)

    const int ka = half * 8;
    const bf16_t* qrow = Qb + ((size_t)h * Sq + m0 + l16) * HD;
    Frag qf[4];
#pragma unroll
    for (int d = 0; d < 4; d++) {
        qf[d].h[0] = *(const v8bf*)(qrow + d * 32 + ka);
        qf[d].h[1] = *(const v8bf*)(qrow + d * 32 + 16 + ka);
    }

    v8f o[8];
    float mi[8], li[8], alpha[8];
#pragma unroll
    for (int t = 0; t < 8; t++) {
        o[t] = (v8f){0.f,0.f,0.f,0.f,0.f,0.f,0.f,0.f};
        mi[t] = -1e30f; li[t] = 0.f;
    }

    const int kb_start = (m0 >= SWIN) ? ((m0 - (SWIN - 1)) & ~31) : 0;
    for (int kb = kb_start; kb <= m0 + 15; kb += 32) {
        // scores: two 16x16 tiles over HD=128 (4 WMMA each)
        v8f c0 = {0.f,0.f,0.f,0.f,0.f,0.f,0.f,0.f};
        v8f c1 = {0.f,0.f,0.f,0.f,0.f,0.f,0.f,0.f};
        {
            const bf16_t* kr0 = Kb + ((size_t)hk * Sq + kb + l16) * HD;
            const bf16_t* kr1 = Kb + ((size_t)hk * Sq + kb + 16 + l16) * HD;
#pragma unroll
            for (int d = 0; d < 4; d++) {
                Frag b0, b1;
                b0.h[0] = *(const v8bf*)(kr0 + d * 32 + ka);
                b0.h[1] = *(const v8bf*)(kr0 + d * 32 + 16 + ka);
                b1.h[0] = *(const v8bf*)(kr1 + d * 32 + ka);
                b1.h[1] = *(const v8bf*)(kr1 + d * 32 + 16 + ka);
                c0 = wmma_bf16(qf[d].v, b0.v, c0);
                c1 = wmma_bf16(qf[d].v, b1.v, c1);
            }
        }
        // softcap + mask + online softmax stats (rows = j + 8*half)
#pragma unroll
        for (int j = 0; j < 8; j++) {
            int row  = m0 + j + 8 * half;
            int col0 = kb + l16;
            int col1 = kb + 16 + l16;
            float x0 = c0[j] * 0.0625f;  x0 = 50.f * tanhf(x0 * 0.02f);
            float x1 = c1[j] * 0.0625f;  x1 = 50.f * tanhf(x1 * 0.02f);
            if (col0 > row || row - col0 >= SWIN) x0 -= 1e9f;
            if (col1 > row || row - col1 >= SWIN) x1 -= 1e9f;
            float rm = half_max16(fmaxf(x0, x1));
            float mn = fmaxf(mi[j], rm);
            float al = __expf(mi[j] - mn);
            float p0 = __expf(x0 - mn);
            float p1 = __expf(x1 - mn);
            float rs = half_sum16(p0 + p1);
            li[j] = li[j] * al + rs;
            mi[j] = mn;
            alpha[j] = al;
            pls[w][j + 8 * half][l16]      = (bf16_t)p0;
            pls[w][j + 8 * half][16 + l16] = (bf16_t)p1;
        }
        asm volatile("s_wait_dscnt 0" ::: "memory");  // LDS in-order per wave

        Frag pf;  // P in A-layout: row = l16, K = keys within block
        pf.h[0] = *(const v8bf*)&pls[w][l16][ka];
        pf.h[1] = *(const v8bf*)&pls[w][l16][16 + ka];

#pragma unroll
        for (int nt = 0; nt < 8; nt++) {
#pragma unroll
            for (int j = 0; j < 8; j++) o[nt][j] *= alpha[j];
            Frag vf;
            const bf16_t* vrow = Vt + ((size_t)hk * HD + nt * 16 + l16) * Sq + kb + ka;
            vf.h[0] = *(const v8bf*)(vrow);
            vf.h[1] = *(const v8bf*)(vrow + 16);
            o[nt] = wmma_bf16(pf.v, vf.v, o[nt]);
        }
    }

    // normalize and write bf16 [S][H*HD]
#pragma unroll
    for (int j = 0; j < 8; j++) {
        float inv = 1.f / li[j];
        int row = m0 + j + 8 * half;
#pragma unroll
        for (int nt = 0; nt < 8; nt++)
            Ab[(size_t)row * (Hq * HD) + h * 128 + nt * 16 + l16] = (bf16_t)(o[nt][j] * inv);
    }
}

// ---------- host launch ----------
extern "C" void kernel_launch(void* const* d_in, const int* in_sizes, int n_in,
                              void* d_out, int out_size, void* d_ws, size_t ws_size,
                              hipStream_t stream) {
    const float* hidden = (const float*)d_in[0];
    const float* cosp   = (const float*)d_in[1];
    const float* sinp   = (const float*)d_in[2];
    // d_in[3] = attention_mask (sliding-window causal, computed analytically)
    const float* Wq = (const float*)d_in[4];
    const float* Wk = (const float*)d_in[5];
    const float* Wv = (const float*)d_in[6];
    const float* Wo = (const float*)d_in[7];
    const float* qnw = (const float*)d_in[8];
    const float* knw = (const float*)d_in[9];
    float* outf = (float*)d_out;

    char* ws = (char*)d_ws;
    size_t off = 0;
    auto take = [&](size_t bytes) -> void* {
        void* p = ws + off;
        off += (bytes + 255) & ~(size_t)255;
        return p;
    };
    bf16_t* Xbf = (bf16_t*)take((size_t)Sq * Dm * 2);
    bf16_t* WqT = (bf16_t*)take((size_t)Dm * (Hq * HD) * 2);
    bf16_t* WkT = (bf16_t*)take((size_t)Dm * (KVh * HD) * 2);
    bf16_t* WvT = (bf16_t*)take((size_t)Dm * (KVh * HD) * 2);
    bf16_t* WoT = (bf16_t*)take((size_t)(Hq * HD) * Dm * 2);
    bf16_t* Qb  = (bf16_t*)take((size_t)Hq  * Sq * HD * 2);
    bf16_t* Kb  = (bf16_t*)take((size_t)KVh * Sq * HD * 2);
    bf16_t* Vt  = (bf16_t*)take((size_t)KVh * Sq * HD * 2);
    bf16_t* Ab  = (bf16_t*)take((size_t)Sq * (Hq * HD) * 2);

    // 1. convert activations + weights to bf16 (weights transposed to [N][K])
    cvt_bf16_kernel<<<2048, 256, 0, stream>>>(hidden, Xbf, (size_t)Sq * Dm);
    transpose_cvt_kernel<<<2048, 256, 0, stream>>>(Wq, WqT, Dm, Hq * HD);
    transpose_cvt_kernel<<<1024, 256, 0, stream>>>(Wk, WkT, Dm, KVh * HD);
    transpose_cvt_kernel<<<1024, 256, 0, stream>>>(Wv, WvT, Dm, KVh * HD);
    transpose_cvt_kernel<<<2048, 256, 0, stream>>>(Wo, WoT, Hq * HD, Dm);

    // 2. QKV projections (rmsnorm+rope fused for Q/K, transposed store for V)
    gemm_head_kernel<0><<<dim3(Sq / 64, Hq),  128, 0, stream>>>(
        Xbf, WqT, Dm, cosp, sinp, qnw, Qb, nullptr, Sq, 0);
    gemm_head_kernel<0><<<dim3(Sq / 64, KVh), 128, 0, stream>>>(
        Xbf, WkT, Dm, cosp, sinp, knw, Kb, nullptr, Sq, 0);
    gemm_head_kernel<1><<<dim3(Sq / 64, KVh), 128, 0, stream>>>(
        Xbf, WvT, Dm, nullptr, nullptr, nullptr, Vt, nullptr, Sq, 0);

    // 3. sliding-window flash attention with tanh softcap
    attn_kernel<<<dim3(Sq / 64, Hq), 128, 0, stream>>>(Qb, Kb, Vt, Ab);

    // 4. output projection -> fp32 d_out
    gemm_head_kernel<2><<<dim3(Sq / 64, Hq), 128, 0, stream>>>(
        Ab, WoT, Hq * HD, nullptr, nullptr, nullptr, nullptr, outf, Sq, Dm);
}
